// SegformerEfficientSelfAttention_29137058136609
// MI455X (gfx1250) — matrix operand order repaired and checked by reading, not compiled, not run on hardware
//
#include <hip/hip_runtime.h>
#include <hip/hip_fp16.h>

typedef __attribute__((ext_vector_type(16))) _Float16 v16h;
typedef __attribute__((ext_vector_type(8)))  float    v8f;
typedef unsigned int u32;

// ---------------------------------------------------------------------------
// CDNA5 WMMA 16x16x32 f16 (wave32).  Fragment-packed storage: each fragment
// is stored as lane-major [32 lanes][16 halves] = 512 halves = 1KB, so a
// fragment load is one aligned 32B vector load (2 x b128).
//
// A 16x32 (MxK): lane = ((k>>3)&1)<<4 | m ; elem = (k&7) + ((k>>4)<<3)
// B 32x16 (KxN): lane = ((k>>3)&1)<<4 | n ; elem = (k&7) + ((k>>4)<<3)
// C/D 16x16 f32: elem i = row (lane>>4)*8 + i, col lane&15
// ---------------------------------------------------------------------------
__device__ __forceinline__ v8f wmma_f16(v16h a, v16h b, v8f c) {
    return __builtin_amdgcn_wmma_f32_16x16x32_f16(
        false, a, false, b, (short)0, c, false, false);
}

__device__ __forceinline__ int a_off(int m, int k) {
    int lane = (((k >> 3) & 1) << 4) | m;
    int elem = (k & 7) + ((k >> 4) << 3);
    return lane * 16 + elem;
}
__device__ __forceinline__ int b_off(int k, int n) {
    return (((((k >> 3) & 1) << 4) | n) << 4) + (k & 7) + ((k >> 4) << 3);
}

__device__ __forceinline__ v16h frag_ld(const _Float16* fragbase, int lane) {
    return *(const v16h*)(fragbase + lane * 16);
}

__device__ __forceinline__ u32 pack2h(float x, float y) {
    union { _Float16 h[2]; u32 u; } u_;
    u_.h[0] = (_Float16)x; u_.h[1] = (_Float16)y;
    return u_.u;
}

// Direct-from-global A fragment (row-major f16 source, stride in halves).
__device__ __forceinline__ v16h gld_a16(const _Float16* base, int stride, int lane, int koff) {
    int m  = lane & 15;
    int kb = ((lane >> 4) << 3) + koff;
    const _Float16* p = base + (size_t)m * stride + kb;
    v16h a;
#pragma unroll
    for (int t = 0; t < 8; ++t) { a[t] = p[t]; a[8 + t] = p[16 + t]; }
    return a;
}

// ---------------------------------------------------------------------------
// Kernel 1: Q projection.  q[b,h,s,d] = sum_c hs[b,s,c]*wq[h*64+d,c] + bq
// ---------------------------------------------------------------------------
__global__ __launch_bounds__(256)
void proj_q_kernel(const float* __restrict__ hs, const float* __restrict__ wq,
                   const float* __restrict__ bq, _Float16* __restrict__ q) {
    const int S = 16384, C = 128;
    int b    = blockIdx.x >> 8;
    int row0 = (blockIdx.x & 255) * 64;
    __shared__ alignas(32) _Float16 Afrag[4 * 512];
    __shared__ alignas(32) _Float16 Bfrag[8 * 512];
    int tid = threadIdx.x, lane = tid & 31, wave = tid >> 5;
    int rb = wave >> 1;
    int nb0 = (wave & 1) * 4;
    v8f acc[4] = {};
    for (int k0 = 0; k0 < 128; k0 += 32) {
        for (int idx = tid; idx < 64 * 16; idx += 256) {
            int r = idx >> 4, kk = (idx & 15) * 2;
            const float* src = &hs[((size_t)b * S + row0 + r) * C + k0 + kk];
            *(u32*)&Afrag[(r >> 4) * 512 + a_off(r & 15, kk)] = pack2h(src[0], src[1]);
        }
        for (int idx = tid; idx < 128 * 16; idx += 256) {
            int n = idx >> 4, kk = (idx & 15) * 2;
            const float* src = &wq[(size_t)n * C + k0 + kk];
            *(u32*)&Bfrag[(n >> 4) * 512 + b_off(kk, n & 15)] = pack2h(src[0], src[1]);
        }
        __syncthreads();
        v16h a = frag_ld(&Afrag[rb * 512], lane);
#pragma unroll
        for (int nb = 0; nb < 4; ++nb)
            acc[nb] = wmma_f16(a, frag_ld(&Bfrag[(nb0 + nb) * 512], lane), acc[nb]);
        __syncthreads();
    }
    int n0 = lane & 15, hi = lane >> 4;
#pragma unroll
    for (int nb = 0; nb < 4; ++nb) {
        int o = (nb0 + nb) * 16 + n0;
        float bias = bq[o];
        int h = o >> 6, d = o & 63;
#pragma unroll
        for (int i = 0; i < 8; ++i) {
            int s = row0 + rb * 16 + hi * 8 + i;
            q[(((size_t)b * 2 + h) * S + s) * 64 + d] = (_Float16)(acc[nb][i] + bias);
        }
    }
}

// ---------------------------------------------------------------------------
// Kernel 2: spatial-reduction conv as im2col GEMM (K = 2048 = c*16+i*4+j).
// ---------------------------------------------------------------------------
__global__ __launch_bounds__(256)
void sr_conv_kernel(const float* __restrict__ hs, const float* __restrict__ srw,
                    const float* __restrict__ srb, float* __restrict__ xr) {
    int b  = blockIdx.x >> 4;
    int p0 = (blockIdx.x & 15) * 64;
    __shared__ alignas(32) _Float16 Afrag[4 * 512];
    __shared__ alignas(32) _Float16 Bfrag[8 * 512];
    int tid = threadIdx.x, lane = tid & 31, wave = tid >> 5;
    int rb = wave >> 1;
    int nb0 = (wave & 1) * 4;
    v8f acc[4] = {};
    for (int k0 = 0; k0 < 2048; k0 += 32) {
        for (int idx = tid; idx < 64 * 16; idx += 256) {
            int r = idx >> 4, kk = (idx & 15) * 2;
            int p = p0 + r, ph = p >> 5, pw = p & 31;
            int k = k0 + kk;
            int c = k >> 4, rem = k & 15, i2 = rem >> 2, j2 = rem & 3;
            int sidx = (4 * ph + i2) * 128 + 4 * pw + j2;
            const float* base = &hs[((size_t)b * 16384 + sidx) * 128 + c];
            *(u32*)&Afrag[(r >> 4) * 512 + a_off(r & 15, kk)] = pack2h(base[0], base[128]);
        }
        for (int idx = tid; idx < 128 * 16; idx += 256) {
            int n = idx >> 4, kk = (idx & 15) * 2;
            const float* src = &srw[(size_t)n * 2048 + k0 + kk];
            *(u32*)&Bfrag[(n >> 4) * 512 + b_off(kk, n & 15)] = pack2h(src[0], src[1]);
        }
        __syncthreads();
        v16h a = frag_ld(&Afrag[rb * 512], lane);
#pragma unroll
        for (int nb = 0; nb < 4; ++nb)
            acc[nb] = wmma_f16(a, frag_ld(&Bfrag[(nb0 + nb) * 512], lane), acc[nb]);
        __syncthreads();
    }
    int n0 = lane & 15, hi = lane >> 4;
#pragma unroll
    for (int nb = 0; nb < 4; ++nb) {
        int o = (nb0 + nb) * 16 + n0;
        float bias = srb[o];
#pragma unroll
        for (int i = 0; i < 8; ++i) {
            int p = p0 + rb * 16 + hi * 8 + i;
            xr[((size_t)b * 1024 + p) * 128 + o] = acc[nb][i] + bias;
        }
    }
}

// ---------------------------------------------------------------------------
// Kernel 3: LayerNorm over C=128, f32 -> f16.
// ---------------------------------------------------------------------------
__global__ __launch_bounds__(128)
void ln_kernel(const float* __restrict__ xr, const float* __restrict__ g,
               const float* __restrict__ beta, _Float16* __restrict__ xln) {
    int row = blockIdx.x;
    int t = threadIdx.x;
    float x = xr[(size_t)row * 128 + t];
    __shared__ float red[8];
    float s = x;
#pragma unroll
    for (int m = 16; m >= 1; m >>= 1) s += __shfl_xor(s, m, 32);
    if ((t & 31) == 0) red[t >> 5] = s;
    __syncthreads();
    float mu = (red[0] + red[1] + red[2] + red[3]) * (1.0f / 128.0f);
    float d0 = x - mu;
    float v2 = d0 * d0;
#pragma unroll
    for (int m = 16; m >= 1; m >>= 1) v2 += __shfl_xor(v2, m, 32);
    if ((t & 31) == 0) red[4 + (t >> 5)] = v2;
    __syncthreads();
    float var = (red[4] + red[5] + red[6] + red[7]) * (1.0f / 128.0f);
    float y = d0 * rsqrtf(var + 1e-5f) * g[t] + beta[t];
    xln[(size_t)row * 128 + t] = (_Float16)y;
}

// ---------------------------------------------------------------------------
// Kernel 4: K/V projection into GLOBAL fragment-packed layouts:
//  mode 0 (K): scores B-frag  B[k=d][n=key];  kpack[bh][kc][dc*2+j][lane][16]
//  mode 1 (V): PV B-frag      B[k=key][n=d];  vpack[bh][kc][nb]    [lane][16]
// ---------------------------------------------------------------------------
__global__ __launch_bounds__(256)
void proj_kv_kernel(const _Float16* __restrict__ xln, const float* __restrict__ w,
                    const float* __restrict__ bias, _Float16* __restrict__ outp,
                    int mode) {
    int b  = blockIdx.x >> 4;
    int p0 = (blockIdx.x & 15) * 64;
    __shared__ alignas(32) _Float16 Afrag[4 * 512];
    __shared__ alignas(32) _Float16 Bfrag[8 * 512];
    int tid = threadIdx.x, lane = tid & 31, wave = tid >> 5;
    int rb = wave >> 1;
    int nb0 = (wave & 1) * 4;
    v8f acc[4] = {};
    for (int k0 = 0; k0 < 128; k0 += 32) {
        for (int idx = tid; idx < 64 * 16; idx += 256) {
            int r = idx >> 4, kk = (idx & 15) * 2;
            u32 p = *(const u32*)&xln[((size_t)b * 1024 + p0 + r) * 128 + k0 + kk];
            *(u32*)&Afrag[(r >> 4) * 512 + a_off(r & 15, kk)] = p;
        }
        for (int idx = tid; idx < 128 * 16; idx += 256) {
            int n = idx >> 4, kk = (idx & 15) * 2;
            const float* src = &w[(size_t)n * 128 + k0 + kk];
            *(u32*)&Bfrag[(n >> 4) * 512 + b_off(kk, n & 15)] = pack2h(src[0], src[1]);
        }
        __syncthreads();
        v16h a = frag_ld(&Afrag[rb * 512], lane);
#pragma unroll
        for (int nb = 0; nb < 4; ++nb)
            acc[nb] = wmma_f16(a, frag_ld(&Bfrag[(nb0 + nb) * 512], lane), acc[nb]);
        __syncthreads();
    }
    int n0 = lane & 15, hi = lane >> 4;
#pragma unroll
    for (int nb = 0; nb < 4; ++nb) {
        int o = (nb0 + nb) * 16 + n0;
        float bv = bias[o];
        int h = o >> 6, d = o & 63;
#pragma unroll
        for (int i = 0; i < 8; ++i) {
            int sp = p0 + rb * 16 + hi * 8 + i;       // key index
            _Float16 val = (_Float16)(acc[nb][i] + bv);
            int bh = b * 2 + h;
            int kc = sp >> 5, kk = sp & 31;
            if (mode == 0) {
                int j = kk >> 4, n = kk & 15;
                int dc = d >> 5, kd = d & 31;
                outp[(((size_t)bh * 32 + kc) * 4 + dc * 2 + j) * 512 + b_off(kd, n)] = val;
            } else {
                int nbv = d >> 4, n = d & 15;
                outp[(((size_t)bh * 32 + kc) * 4 + nbv) * 512 + b_off(kk, n)] = val;
            }
        }
    }
}

// ---------------------------------------------------------------------------
// Kernel 5: flash attention, 64-key chunks.  8 waves x 16 queries per block.
// K/V fragments stream from L2 (fragment-packed global), no block barriers.
// Softmax denominator accumulated by a "ones" WMMA (P x 1 = row sums), so the
// only cross-lane shuffles left are the 4-step row-max reductions.
// Base-2 softmax: scores pre-scaled by 0.125*log2(e), exponentials are bare
// v_exp_f32 via exp2f.
// ---------------------------------------------------------------------------
__global__ __launch_bounds__(256)
void attn_kernel(const _Float16* __restrict__ q, const _Float16* __restrict__ kpack,
                 const _Float16* __restrict__ vpack, float* __restrict__ out) {
    const int S = 16384;
    int bh = blockIdx.x >> 7;
    int qt = blockIdx.x & 127;
    int b = bh >> 1, h = bh & 1;
    int tid = threadIdx.x, lane = tid & 31, wave = tid >> 5;
    int qrow0 = qt * 128 + wave * 16;

    __shared__ alignas(32) _Float16 Psh[8][1024];   // 2 probs tiles per wave

    const _Float16* qbase = q + ((size_t)bh * S + qrow0) * 64;
    v16h aq0 = gld_a16(qbase, 64, lane, 0);
    v16h aq1 = gld_a16(qbase, 64, lane, 32);

    v16h ones;
#pragma unroll
    for (int t = 0; t < 16; ++t) ones[t] = (_Float16)1.0f;

    float mrow[8];
#pragma unroll
    for (int i = 0; i < 8; ++i) mrow[i] = -1e30f;
    v8f o0 = {}, o1 = {}, o2 = {}, o3 = {}, o4 = {};   // o4 = row sums (l)

    const _Float16* kp = kpack + (size_t)bh * 32 * 2048;
    const _Float16* vp = vpack + (size_t)bh * 32 * 2048;
    const float sc2 = 0.125f * 1.44269504088896340736f;   // 1/sqrt(64) * log2(e)
    int n0 = lane & 15, hi = lane >> 4;

    for (int kc = 0; kc < 32; kc += 2) {     // 64 keys per iteration
        const _Float16* kcb = kp + (size_t)kc * 2048;
        const _Float16* vcb = vp + (size_t)kc * 2048;
        if (kc + 2 < 32) {                   // global_prefetch_b8 next chunk
            __builtin_prefetch(kcb + 4096, 0, 1);
            __builtin_prefetch(vcb + 4096, 0, 1);
        }

        // ---- scores: 16x64, f32 accum ----
        v8f s0 = {}, s1 = {}, s2 = {}, s3 = {};
        s0 = wmma_f16(aq0, frag_ld(kcb + 0 * 512, lane), s0);
        s0 = wmma_f16(aq1, frag_ld(kcb + 2 * 512, lane), s0);
        s1 = wmma_f16(aq0, frag_ld(kcb + 1 * 512, lane), s1);
        s1 = wmma_f16(aq1, frag_ld(kcb + 3 * 512, lane), s1);
        s2 = wmma_f16(aq0, frag_ld(kcb + 4 * 512, lane), s2);
        s2 = wmma_f16(aq1, frag_ld(kcb + 6 * 512, lane), s2);
        s3 = wmma_f16(aq0, frag_ld(kcb + 5 * 512, lane), s3);
        s3 = wmma_f16(aq1, frag_ld(kcb + 7 * 512, lane), s3);

        // ---- online softmax (base-2; row groups = 16-lane halves) ----
        float scl[8];
#pragma unroll
        for (int i = 0; i < 8; ++i) {
            float t0 = s0[i] * sc2, t1 = s1[i] * sc2;
            float t2 = s2[i] * sc2, t3 = s3[i] * sc2;
            float mt = fmaxf(fmaxf(t0, t1), fmaxf(t2, t3));
#pragma unroll
            for (int m = 8; m >= 1; m >>= 1) mt = fmaxf(mt, __shfl_xor(mt, m, 32));
            float mn = fmaxf(mrow[i], mt);
            scl[i] = exp2f(mrow[i] - mn);
            mrow[i] = mn;
            float e0 = exp2f(t0 - mn), e1 = exp2f(t1 - mn);
            float e2 = exp2f(t2 - mn), e3 = exp2f(t3 - mn);
            int r = hi * 8 + i;
            Psh[wave][a_off(r, n0)]            = (_Float16)e0;
            Psh[wave][a_off(r, 16 + n0)]       = (_Float16)e1;
            Psh[wave][512 + a_off(r, n0)]      = (_Float16)e2;
            Psh[wave][512 + a_off(r, 16 + n0)] = (_Float16)e3;
        }
#pragma unroll
        for (int i = 0; i < 8; ++i) {
            o0[i] *= scl[i]; o1[i] *= scl[i]; o2[i] *= scl[i]; o3[i] *= scl[i];
            o4[i] *= scl[i];
        }

        asm volatile("s_wait_dscnt 0x0" ::: "memory");   // in-wave LDS RAW
        v16h pa0 = frag_ld(&Psh[wave][0],   lane);
        v16h pa1 = frag_ld(&Psh[wave][512], lane);

        // ---- PV: 16x64 accumulate + row-sum via ones column ----
        o0 = wmma_f16(pa0, frag_ld(vcb + 0 * 512, lane), o0);
        o1 = wmma_f16(pa0, frag_ld(vcb + 1 * 512, lane), o1);
        o2 = wmma_f16(pa0, frag_ld(vcb + 2 * 512, lane), o2);
        o3 = wmma_f16(pa0, frag_ld(vcb + 3 * 512, lane), o3);
        o4 = wmma_f16(pa0, ones, o4);
        o0 = wmma_f16(pa1, frag_ld(vcb + 4 * 512, lane), o0);
        o1 = wmma_f16(pa1, frag_ld(vcb + 5 * 512, lane), o1);
        o2 = wmma_f16(pa1, frag_ld(vcb + 6 * 512, lane), o2);
        o3 = wmma_f16(pa1, frag_ld(vcb + 7 * 512, lane), o3);
        o4 = wmma_f16(pa1, ones, o4);
    }

#pragma unroll
    for (int i = 0; i < 8; ++i) {
        float inv = 1.0f / o4[i];
        o0[i] *= inv; o1[i] *= inv; o2[i] *= inv; o3[i] *= inv;
    }
#pragma unroll
    for (int i = 0; i < 8; ++i) {
        int s = qrow0 + hi * 8 + i;
        size_t base = ((size_t)b * 16384 + s) * 128 + h * 64;
        out[base + n0]      = o0[i];
        out[base + 16 + n0] = o1[i];
        out[base + 32 + n0] = o2[i];
        out[base + 48 + n0] = o3[i];
    }
}

// ---------------------------------------------------------------------------
extern "C" void kernel_launch(void* const* d_in, const int* in_sizes, int n_in,
                              void* d_out, int out_size, void* d_ws, size_t ws_size,
                              hipStream_t stream) {
    const float* hs  = (const float*)d_in[0];
    const float* wq  = (const float*)d_in[1];
    const float* bq  = (const float*)d_in[2];
    const float* wk  = (const float*)d_in[3];
    const float* bk  = (const float*)d_in[4];
    const float* wv  = (const float*)d_in[5];
    const float* bv  = (const float*)d_in[6];
    const float* srw = (const float*)d_in[7];
    const float* srb = (const float*)d_in[8];
    const float* lng = (const float*)d_in[9];
    const float* lnb = (const float*)d_in[10];
    float* out = (float*)d_out;

    char* ws = (char*)d_ws;
    float*    xr    = (float*)ws;                                   // 2 MB
    _Float16* xln   = (_Float16*)(ws + (size_t)2  * 1024 * 1024);   // 1 MB
    _Float16* qh    = (_Float16*)(ws + (size_t)3  * 1024 * 1024);   // 16 MB
    _Float16* kpack = (_Float16*)(ws + (size_t)19 * 1024 * 1024);   // 1 MB
    _Float16* vpack = (_Float16*)(ws + (size_t)20 * 1024 * 1024);   // 1 MB

    proj_q_kernel <<<1024, 256, 0, stream>>>(hs, wq, bq, qh);
    sr_conv_kernel<<<64,   256, 0, stream>>>(hs, srw, srb, xr);
    ln_kernel     <<<4096, 128, 0, stream>>>(xr, lng, lnb, xln);
    proj_kv_kernel<<<64,   256, 0, stream>>>(xln, wk, bk, kpack, 0);
    proj_kv_kernel<<<64,   256, 0, stream>>>(xln, wv, bv, vpack, 1);
    attn_kernel   <<<1024, 256, 0, stream>>>(qh, kpack, vpack, out);
}
